// CausalSelfAttention_30897994728322
// MI455X (gfx1250) — compile-verified, matrix operand order checked by reading, not run on hardware
//
#include <hip/hip_runtime.h>
#include <hip/hip_bf16.h>

// ---------------------------------------------------------------------------
// CausalSelfAttention on MI455X (gfx1250).
// bf16 WMMA pipeline + Tensor Data Mover (TDM) LDS staging for the GEMMs and
// for the flash-attention K/V tiles. B=4, T=2048, C=1024, H=16, D=64.
// ---------------------------------------------------------------------------

typedef __bf16 bf16_t;
typedef __attribute__((ext_vector_type(16))) __bf16 v16bf;
typedef __attribute__((ext_vector_type(8)))  __bf16 v8bf;
typedef __attribute__((ext_vector_type(8)))  float  v8f;
typedef __attribute__((ext_vector_type(4)))  unsigned int u32x4;
typedef __attribute__((ext_vector_type(8)))  int i32x8;
typedef __attribute__((ext_vector_type(4)))  int i32x4;

#define B_   4
#define T_   2048
#define C_   1024
#define H_   16
#define D_   64
#define MROWS (B_ * T_)      // 8192
#define NQKV  (3 * C_)       // 3072

__device__ __forceinline__ v8f zero8() {
  v8f z = {0.f, 0.f, 0.f, 0.f, 0.f, 0.f, 0.f, 0.f};
  return z;
}

// Load one 16x32 bf16 WMMA operand fragment from a row-major matrix (global
// or LDS). Per CDNA5 ISA 16-bit A layout: lanes 0-15 -> K in {kbase+0..7,
// kbase+16..23}, lanes 16-31 -> K in {kbase+8..15, kbase+24..31}. Row index
// must already include (lane & 15).
__device__ __forceinline__ v16bf load_frag_rm(const bf16_t* __restrict__ base,
                                              int row, int ld, int kbase) {
  const int lane = threadIdx.x & 31;
  const int klo  = (lane < 16) ? 0 : 8;
  const bf16_t* p = base + (size_t)row * (size_t)ld + (size_t)(kbase + klo);
  v8bf lo = *(const v8bf*)(p);
  v8bf hi = *(const v8bf*)(p + 16);
  return __builtin_shufflevector(lo, hi, 0, 1, 2, 3, 4, 5, 6, 7,
                                         8, 9, 10, 11, 12, 13, 14, 15);
}

__device__ __forceinline__ v8f wmma_bf16(v16bf a, v16bf b, v8f c) {
  return __builtin_amdgcn_wmma_f32_16x16x32_bf16(false, a, false, b,
                                                 (short)0, c, false, false);
}

// ---------------------------------------------------------------------------
// TDM: DMA a 2D bf16 tile [tile_rows x tile_k] (row stride = ld elements)
// from global memory into LDS at byte offset lds_off (rows stored
// contiguously: LDS layout [tile_rows][tile_k]). Descriptor packing per
// CDNA5 ISA section 8.3/8.4. Issued once per calling wave; tracked by
// TENSORcnt. EXEC is ignored by tensor ops.
// ---------------------------------------------------------------------------
__device__ __forceinline__ void tdm_load_tile_2d(unsigned int lds_off,
                                                 const bf16_t* gsrc,
                                                 unsigned int tile_k,
                                                 unsigned int tile_rows,
                                                 unsigned int ld,
                                                 unsigned int tensor_rows) {
  const unsigned long long ga = (unsigned long long)(uintptr_t)gsrc;
  u32x4 g0;
  g0[0] = 1u;                                   // count=1 (valid user D#)
  g0[1] = lds_off;                              // LDS byte address
  g0[2] = (unsigned int)ga;                     // global addr [31:0]
  g0[3] = (unsigned int)((ga >> 32) & 0x1FFFFFFu) | (2u << 30);  // [56:32]|type=2

  const unsigned int td0 = ld;                  // tensor_dim0 (elems)
  const unsigned int td1 = tensor_rows;         // tensor_dim1 (rows)
  i32x8 g1;
  g1[0] = (int)(1u << 16);                      // data_size=1 (2-byte elems)
  g1[1] = (int)((td0 & 0xFFFFu) << 16);         // tensor_dim0[15:0]
  g1[2] = (int)(((td0 >> 16) & 0xFFFFu) | ((td1 & 0xFFFFu) << 16));
  g1[3] = (int)(((td1 >> 16) & 0xFFFFu) | ((tile_k & 0xFFFFu) << 16)); // tile_dim0
  g1[4] = (int)(tile_rows & 0xFFFFu);           // tile_dim1 (tile_dim2=0)
  g1[5] = (int)ld;                              // tensor_dim0_stride[31:0]
  g1[6] = 0;                                    // stride0[47:32], stride1[15:0]
  g1[7] = 0;

  i32x4 z4 = {0, 0, 0, 0};
#if __has_include(<hip/amd_detail/amd_gfx1250_TDM.h>)
  i32x8 z8 = {0, 0, 0, 0, 0, 0, 0, 0};
  __builtin_amdgcn_tensor_load_to_lds(g0, g1, z4, z4, z8, 0);
#else
  __builtin_amdgcn_tensor_load_to_lds(g0, g1, z4, z4, 0);
#endif
}

// ---------------------------------------------------------------------------
// fp32 -> bf16 elementwise convert
// ---------------------------------------------------------------------------
__global__ void cvt_f32_bf16_kernel(const float* __restrict__ in,
                                    bf16_t* __restrict__ out, int n) {
  int i = blockIdx.x * blockDim.x + threadIdx.x;
  if (i < n) out[i] = (bf16_t)in[i];
}

// ---------------------------------------------------------------------------
// Shared TDM-staged GEMM core: block tile 128(M) x 128(N), K-step 32,
// 8 waves (4 M-waves x 2 N-waves), each wave 32x64 via 2x4 WMMA tiles.
// A [Mtot x K], W [Ntot x K] row-major bf16; acc = A * W^T (fp32).
// Double-buffered LDS staging via tensor_load_to_lds issued by wave 0.
// ---------------------------------------------------------------------------
#define GEMM_CORE(A_, W_, K_, bm_, bn_, acc_)                                 \
  __shared__ __align__(16) bf16_t tA[2][128 * 32];                            \
  __shared__ __align__(16) bf16_t tB[2][128 * 32];                            \
  const int lane = threadIdx.x & 31;                                          \
  const int wid  = threadIdx.x >> 5;                                          \
  const int wm   = wid >> 1;                                                  \
  const int wn   = wid & 1;                                                   \
  v8f acc_[2][4];                                                             \
  _Pragma("unroll") for (int mi = 0; mi < 2; ++mi)                            \
      _Pragma("unroll") for (int ni = 0; ni < 4; ++ni) acc_[mi][ni] = zero8();\
  if (wid == 0) {                                                             \
    tdm_load_tile_2d((unsigned int)(uintptr_t)&tA[0][0],                      \
                     A_ + (size_t)(bm_) * 128 * (K_), 32u, 128u,              \
                     (unsigned int)(K_), 128u);                               \
    tdm_load_tile_2d((unsigned int)(uintptr_t)&tB[0][0],                      \
                     W_ + (size_t)(bn_) * 128 * (K_), 32u, 128u,              \
                     (unsigned int)(K_), 128u);                               \
  }                                                                           \
  for (int k0 = 0; k0 < (K_); k0 += 32) {                                     \
    const int buf = (k0 >> 5) & 1;                                            \
    if (wid == 0) __builtin_amdgcn_s_wait_tensorcnt(0);                       \
    __syncthreads();                                                          \
    if (wid == 0 && (k0 + 32) < (K_)) {                                       \
      tdm_load_tile_2d((unsigned int)(uintptr_t)&tA[buf ^ 1][0],              \
                       A_ + (size_t)(bm_) * 128 * (K_) + (k0 + 32), 32u,      \
                       128u, (unsigned int)(K_), 128u);                       \
      tdm_load_tile_2d((unsigned int)(uintptr_t)&tB[buf ^ 1][0],              \
                       W_ + (size_t)(bn_) * 128 * (K_) + (k0 + 32), 32u,      \
                       128u, (unsigned int)(K_), 128u);                       \
    }                                                                         \
    v16bf a0 = load_frag_rm(&tA[buf][0], wm * 32 + (lane & 15), 32, 0);       \
    v16bf a1 = load_frag_rm(&tA[buf][0], wm * 32 + 16 + (lane & 15), 32, 0);  \
    _Pragma("unroll") for (int ni = 0; ni < 4; ++ni) {                        \
      v16bf b = load_frag_rm(&tB[buf][0],                                     \
                             wn * 64 + ni * 16 + (lane & 15), 32, 0);         \
      acc_[0][ni] = wmma_bf16(a0, b, acc_[0][ni]);                            \
      acc_[1][ni] = wmma_bf16(a1, b, acc_[1][ni]);                            \
    }                                                                         \
    __syncthreads();                                                          \
  }

// ---------------------------------------------------------------------------
// QKV GEMM: qkv[m, o] = sum_k x[m,k] * w_attn[o,k] + b_attn[o]
// Epilogue scatters q (scaled 1/sqrt(D)) -> [B,H,T,D], k -> [B,H,T,D],
// v transposed -> [B,H,D,T], all bf16.
// ---------------------------------------------------------------------------
__global__ __launch_bounds__(256) void qkv_gemm_kernel(
    const bf16_t* __restrict__ A,   // [MROWS, C_]
    const bf16_t* __restrict__ Wm,  // [NQKV, C_]
    const float*  __restrict__ bias,
    bf16_t* __restrict__ qb, bf16_t* __restrict__ kb, bf16_t* __restrict__ vt) {
  const int bn = blockIdx.x % (NQKV / 128);   // 24
  const int bm = blockIdx.x / (NQKV / 128);   // 64
  GEMM_CORE(A, Wm, C_, bm, bn, acc)

  const int m0 = bm * 128 + wm * 32;
  const int n0 = bn * 128 + wn * 64;
#pragma unroll
  for (int mi = 0; mi < 2; ++mi)
#pragma unroll
    for (int ni = 0; ni < 4; ++ni) {
      const int col = n0 + ni * 16 + (lane & 15);
      const float bv = bias[col];
#pragma unroll
      for (int v = 0; v < 8; ++v) {
        const int row = m0 + mi * 16 + ((lane >> 4) << 3) + v;
        const float val = acc[mi][ni][v] + bv;
        const int bb = row >> 11;        // row / T_
        const int t  = row & (T_ - 1);
        if (col < C_) {
          const int h = col >> 6, d = col & 63;
          qb[((size_t)(bb * H_ + h) * T_ + t) * D_ + d] = (bf16_t)(val * 0.125f);
        } else if (col < 2 * C_) {
          const int c2 = col - C_;
          const int h = c2 >> 6, d = c2 & 63;
          kb[((size_t)(bb * H_ + h) * T_ + t) * D_ + d] = (bf16_t)val;
        } else {
          const int c2 = col - 2 * C_;
          const int h = c2 >> 6, d = c2 & 63;
          vt[((size_t)(bb * H_ + h) * D_ + d) * T_ + t] = (bf16_t)val;
        }
      }
    }
}

// ---------------------------------------------------------------------------
// Output projection: out[m, o] = sum_k y[m,k] * w_proj[o,k] + b_proj[o]
// ---------------------------------------------------------------------------
__global__ __launch_bounds__(256) void proj_gemm_kernel(
    const bf16_t* __restrict__ A,   // [MROWS, C_]
    const bf16_t* __restrict__ Wm,  // [C_, C_]
    const float*  __restrict__ bias,
    float* __restrict__ out) {
  const int bn = blockIdx.x % (C_ / 128);     // 8
  const int bm = blockIdx.x / (C_ / 128);     // 64
  GEMM_CORE(A, Wm, C_, bm, bn, acc)

  const int m0 = bm * 128 + wm * 32;
  const int n0 = bn * 128 + wn * 64;
#pragma unroll
  for (int mi = 0; mi < 2; ++mi)
#pragma unroll
    for (int ni = 0; ni < 4; ++ni) {
      const int col = n0 + ni * 16 + (lane & 15);
      const float bv = bias[col];
#pragma unroll
      for (int v = 0; v < 8; ++v) {
        const int row = m0 + mi * 16 + ((lane >> 4) << 3) + v;
        out[(size_t)row * C_ + col] = acc[mi][ni][v] + bv;
      }
    }
}

// ---------------------------------------------------------------------------
// Flash-style causal attention with TDM-staged K/V tiles.
// Block = 8 waves handling 8 consecutive 16-row q-tiles of the SAME head.
// Per 64-key block: K tile (64x64) and V^T tile (64x64) are DMA'd into LDS
// (double-buffered, 32 KB) by wave 0 and shared by all 8 waves, overlapping
// the previous block's WMMAs. Q pre-scaled by 1/sqrt(D). Output y[B,T,C] bf16.
// ---------------------------------------------------------------------------
__global__ __launch_bounds__(256) void flash_attn_kernel(
    const bf16_t* __restrict__ qb, const bf16_t* __restrict__ kb,
    const bf16_t* __restrict__ vt, bf16_t* __restrict__ yb) {
  __shared__ __align__(16) bf16_t ktile[2][64 * 64];  // [key][d]
  __shared__ __align__(16) bf16_t vtile[2][64 * 64];  // [d][key]
  __shared__ __align__(16) bf16_t ptile[8][16 * 64];
  const int lane = threadIdx.x & 31;
  const int wv   = threadIdx.x >> 5;
  const int gw   = (blockIdx.x * blockDim.x + threadIdx.x) >> 5;
  const int qt = gw & (T_ / 16 - 1);      // 0..127
  const int hh = (gw >> 7) & (H_ - 1);
  const int bb = gw >> 11;
  if (bb >= B_) return;

  const bf16_t* qh = qb + (size_t)(bb * H_ + hh) * T_ * D_;
  const bf16_t* kh = kb + (size_t)(bb * H_ + hh) * T_ * D_;
  const bf16_t* vh = vt + (size_t)(bb * H_ + hh) * D_ * T_;

  v16bf aq0 = load_frag_rm(qh, qt * 16 + (lane & 15), D_, 0);
  v16bf aq1 = load_frag_rm(qh, qt * 16 + (lane & 15), D_, 32);

  float mrow[8], lrow[8];
  v8f o[4];
#pragma unroll
  for (int v = 0; v < 8; ++v) { mrow[v] = -__builtin_inff(); lrow[v] = 0.f; }
#pragma unroll
  for (int ni = 0; ni < 4; ++ni) o[ni] = zero8();

  const int my_nkb  = qt / 4 + 1;              // per-wave causal block count
  const int nkb_max = (qt & ~7) / 4 + 2;       // block-wide max (8 q-tiles)

  if (wv == 0) {
    tdm_load_tile_2d((unsigned int)(uintptr_t)&ktile[0][0], kh, 64u, 64u,
                     (unsigned int)D_, 64u);
    tdm_load_tile_2d((unsigned int)(uintptr_t)&vtile[0][0], vh, 64u, 64u,
                     (unsigned int)T_, 64u);
  }

  for (int blk = 0; blk < nkb_max; ++blk) {
    const int s0 = blk * 64;
    const int buf = blk & 1;
    if (wv == 0) __builtin_amdgcn_s_wait_tensorcnt(0);
    __syncthreads();
    if (wv == 0 && (blk + 1) < nkb_max) {
      tdm_load_tile_2d((unsigned int)(uintptr_t)&ktile[buf ^ 1][0],
                       kh + (size_t)(s0 + 64) * D_, 64u, 64u,
                       (unsigned int)D_, 64u);
      tdm_load_tile_2d((unsigned int)(uintptr_t)&vtile[buf ^ 1][0],
                       vh + (s0 + 64), 64u, 64u,
                       (unsigned int)T_, 64u);
    }

    if (blk < my_nkb) {
      v8f s[4];
#pragma unroll
      for (int ni = 0; ni < 4; ++ni) s[ni] = zero8();

      // S = Q * K^T (contraction over D=64 -> two K=32 steps), K from LDS
#pragma unroll
      for (int ni = 0; ni < 4; ++ni) {
        v16bf b0 = load_frag_rm(&ktile[buf][0], ni * 16 + (lane & 15), 64, 0);
        v16bf b1 = load_frag_rm(&ktile[buf][0], ni * 16 + (lane & 15), 64, 32);
        s[ni] = wmma_bf16(aq0, b0, s[ni]);
        s[ni] = wmma_bf16(aq1, b1, s[ni]);
      }

      if (blk == my_nkb - 1) {  // diagonal block: causal mask
#pragma unroll
        for (int ni = 0; ni < 4; ++ni)
#pragma unroll
          for (int v = 0; v < 8; ++v) {
            const int colg = s0 + ni * 16 + (lane & 15);
            const int rowg = qt * 16 + ((lane >> 4) << 3) + v;
            if (colg > rowg) s[ni][v] = -1e30f;
          }
      }

      // online softmax row statistics (rows live across 16-lane halves)
      float mnew[8], alpha[8];
#pragma unroll
      for (int v = 0; v < 8; ++v) {
        float mx = mrow[v];
#pragma unroll
        for (int ni = 0; ni < 4; ++ni) mx = fmaxf(mx, s[ni][v]);
        mx = fmaxf(mx, __shfl_xor(mx, 1));
        mx = fmaxf(mx, __shfl_xor(mx, 2));
        mx = fmaxf(mx, __shfl_xor(mx, 4));
        mx = fmaxf(mx, __shfl_xor(mx, 8));
        mnew[v] = mx;
        alpha[v] = __expf(mrow[v] - mx);
        mrow[v] = mx;
      }
#pragma unroll
      for (int ni = 0; ni < 4; ++ni)
#pragma unroll
        for (int v = 0; v < 8; ++v) s[ni][v] = __expf(s[ni][v] - mnew[v]);
#pragma unroll
      for (int v = 0; v < 8; ++v) {
        float rs = s[0][v] + s[1][v] + s[2][v] + s[3][v];
        rs += __shfl_xor(rs, 1);
        rs += __shfl_xor(rs, 2);
        rs += __shfl_xor(rs, 4);
        rs += __shfl_xor(rs, 8);
        lrow[v] = lrow[v] * alpha[v] + rs;
      }
#pragma unroll
      for (int ni = 0; ni < 4; ++ni)
#pragma unroll
        for (int v = 0; v < 8; ++v) o[ni][v] *= alpha[v];

      // Relayout P from C-layout to A-layout through LDS (2 KB per wave).
#pragma unroll
      for (int ni = 0; ni < 4; ++ni)
#pragma unroll
        for (int v = 0; v < 8; ++v) {
          const int r = ((lane >> 4) << 3) + v;
          const int c = ni * 16 + (lane & 15);
          ptile[wv][r * 64 + c] = (bf16_t)s[ni][v];
        }
      asm volatile("s_wait_dscnt 0" ::: "memory");

      v16bf ap0 = load_frag_rm(&ptile[wv][0], lane & 15, 64, 0);
      v16bf ap1 = load_frag_rm(&ptile[wv][0], lane & 15, 64, 32);

      // O += P * V (contraction over s), V^T tile [d][key] from LDS
#pragma unroll
      for (int ni = 0; ni < 4; ++ni) {
        v16bf bv0 = load_frag_rm(&vtile[buf][0], ni * 16 + (lane & 15), 64, 0);
        v16bf bv1 = load_frag_rm(&vtile[buf][0], ni * 16 + (lane & 15), 64, 32);
        o[ni] = wmma_bf16(ap0, bv0, o[ni]);
        o[ni] = wmma_bf16(ap1, bv1, o[ni]);
      }
    }
    __syncthreads();
  }

  // normalize and write y[B,T,C] bf16
#pragma unroll
  for (int ni = 0; ni < 4; ++ni)
#pragma unroll
    for (int v = 0; v < 8; ++v) {
      const int t = qt * 16 + ((lane >> 4) << 3) + v;
      const int d = ni * 16 + (lane & 15);
      const float val = o[ni][v] / lrow[v];
      yb[((size_t)bb * T_ + t) * C_ + hh * 64 + d] = (bf16_t)val;
    }
}

// ---------------------------------------------------------------------------
// Host launcher
// ---------------------------------------------------------------------------
static inline size_t align256(size_t x) { return (x + 255) & ~(size_t)255; }

extern "C" void kernel_launch(void* const* d_in, const int* in_sizes, int n_in,
                              void* d_out, int out_size, void* d_ws,
                              size_t ws_size, hipStream_t stream) {
  const float* x      = (const float*)d_in[0];  // [B,T,C]
  const float* w_attn = (const float*)d_in[1];  // [3C,C]
  const float* b_attn = (const float*)d_in[2];  // [3C]
  const float* w_proj = (const float*)d_in[3];  // [C,C]
  const float* b_proj = (const float*)d_in[4];  // [C]
  float* out = (float*)d_out;                   // [B,T,C]

  char* ws = (char*)d_ws;
  size_t off = 0;
  bf16_t* xb = (bf16_t*)(ws + off); off = align256(off + (size_t)MROWS * C_ * 2);
  bf16_t* wa = (bf16_t*)(ws + off); off = align256(off + (size_t)NQKV * C_ * 2);
  bf16_t* wp = (bf16_t*)(ws + off); off = align256(off + (size_t)C_ * C_ * 2);
  bf16_t* qb = (bf16_t*)(ws + off); off = align256(off + (size_t)B_ * H_ * T_ * D_ * 2);
  bf16_t* kb = (bf16_t*)(ws + off); off = align256(off + (size_t)B_ * H_ * T_ * D_ * 2);
  bf16_t* vt = (bf16_t*)(ws + off); off = align256(off + (size_t)B_ * H_ * T_ * D_ * 2);
  bf16_t* yb = (bf16_t*)(ws + off); off = align256(off + (size_t)MROWS * C_ * 2);

  const int nx  = MROWS * C_;     // 8,388,608
  const int nwa = NQKV * C_;      // 3,145,728
  const int nwp = C_ * C_;        // 1,048,576
  cvt_f32_bf16_kernel<<<(nx + 255) / 256, 256, 0, stream>>>(x, xb, nx);
  cvt_f32_bf16_kernel<<<(nwa + 255) / 256, 256, 0, stream>>>(w_attn, wa, nwa);
  cvt_f32_bf16_kernel<<<(nwp + 255) / 256, 256, 0, stream>>>(w_proj, wp, nwp);

  // QKV: (MROWS/128) * (NQKV/128) = 64 * 24 = 1536 block tiles
  qkv_gemm_kernel<<<1536, 256, 0, stream>>>(xb, wa, b_attn, qb, kb, vt);

  // Flash attention: B*H*(T/16) = 8192 q-tiles, 8 per block (same head)
  flash_attn_kernel<<<8192 / 8, 256, 0, stream>>>(qb, kb, vt, yb);

  // Projection: (MROWS/128) * (C_/128) = 64 * 8 = 512 block tiles
  proj_gemm_kernel<<<512, 256, 0, stream>>>(yb, wp, b_proj, out);
}
